// GNN_21165598835487
// MI455X (gfx1250) — matrix operand order repaired and checked by reading, not compiled, not run on hardware
//
#include <hip/hip_runtime.h>
#include <math.h>

#define N_NODES 3072
#define F_DIM   512
#define CLS     40
#define CLS_PAD 64
#define NE      61440
#define KCAT    6144   /* 2*N_NODES */
#define HK      1024   /* 2*F_DIM   */

typedef _Float16 h16;
typedef __attribute__((ext_vector_type(16))) _Float16 v16h;
typedef __attribute__((ext_vector_type(8)))  _Float16 v8h;
typedef __attribute__((ext_vector_type(8)))  float    v8f;

// ---------------------------------------------------------------------------
// WMMA GEMM:  C[M,N](f32) = A[M,K](f16, row-major) @ Bt[N,K](f16, row-major)^T
// grid = (M/128, N/64), block = 256 (8 waves); wave -> 32x32 tile (2x2 WMMA)
// ---------------------------------------------------------------------------
__device__ __forceinline__ v16h load_fragA(const h16* p) {
  v8h lo = *(const v8h*)p;          // K = base+0..7
  v8h hi = *(const v8h*)(p + 16);   // K = base+16..23
  return __builtin_shufflevector(lo, hi, 0,1,2,3,4,5,6,7,8,9,10,11,12,13,14,15);
}
__device__ __forceinline__ v16h load_fragB(const h16* p) {
  v8h lo = *(const v8h*)p;          // K = base+0..7
  v8h hi = *(const v8h*)(p + 16);   // K = base+16..23
  return __builtin_shufflevector(lo, hi, 0,1,2,3,4,5,6,7,8,9,10,11,12,13,14,15);
}

__global__ void __launch_bounds__(256) gemm_tn_f16f32(
    const h16* __restrict__ A, const h16* __restrict__ Bt,
    float* __restrict__ C, int lda, int ldb, int ldc, int K)
{
  const int lane = threadIdx.x & 31;
  const int wid  = threadIdx.x >> 5;
  const int tm   = blockIdx.x * 128 + (wid & 3) * 32;
  const int tn   = blockIdx.y * 64  + (wid >> 2) * 32;
  const int hlf  = lane >> 4;     // lane half selects K sub-range per ISA layout
  const int l16  = lane & 15;

  const h16* a0 = A  + (size_t)(tm +      l16) * lda + hlf * 8;
  const h16* a1 = A  + (size_t)(tm + 16 + l16) * lda + hlf * 8;
  const h16* b0 = Bt + (size_t)(tn +      l16) * ldb + hlf * 8;
  const h16* b1 = Bt + (size_t)(tn + 16 + l16) * ldb + hlf * 8;

  v8f c00 = {}, c01 = {}, c10 = {}, c11 = {};
  for (int k = 0; k < K; k += 32) {
    v16h fa0 = load_fragA(a0 + k);
    v16h fa1 = load_fragA(a1 + k);
    v16h fb0 = load_fragB(b0 + k);
    v16h fb1 = load_fragB(b1 + k);
    c00 = __builtin_amdgcn_wmma_f32_16x16x32_f16(false, fa0, false, fb0, (short)0, c00, false, false);
    c01 = __builtin_amdgcn_wmma_f32_16x16x32_f16(false, fa0, false, fb1, (short)0, c01, false, false);
    c10 = __builtin_amdgcn_wmma_f32_16x16x32_f16(false, fa1, false, fb0, (short)0, c10, false, false);
    c11 = __builtin_amdgcn_wmma_f32_16x16x32_f16(false, fa1, false, fb1, (short)0, c11, false, false);
  }
  const int r0 = hlf * 8;
  #pragma unroll
  for (int j = 0; j < 8; ++j) {
    C[(size_t)(tm + r0 + j)      * ldc + tn + l16]      = c00[j];
    C[(size_t)(tm + r0 + j)      * ldc + tn + 16 + l16] = c01[j];
    C[(size_t)(tm + 16 + r0 + j) * ldc + tn + l16]      = c10[j];
    C[(size_t)(tm + 16 + r0 + j) * ldc + tn + 16 + l16] = c11[j];
  }
}

// ---------------------------------------------------------------------------
// Support kernels
// ---------------------------------------------------------------------------
__global__ void zero_f32(float* p, size_t n) {
  size_t i = (size_t)blockIdx.x * 256 + threadIdx.x;
  if (i < n) p[i] = 0.0f;
}

__global__ void scatter_edges(const int* __restrict__ ei, float* __restrict__ A) {
  int e = blockIdx.x * 256 + threadIdx.x;
  if (e >= NE) return;
  int r = ei[e], c = ei[NE + e];
  atomicAdd(&A[(size_t)r * N_NODES + c], 1.0f);
}

__global__ void rowsum_k(const float* __restrict__ M, float* __restrict__ out, int ncols) {
  __shared__ float sm[256];
  int row = blockIdx.x;
  float s = 0.0f;
  for (int j = threadIdx.x; j < ncols; j += 256) s += M[(size_t)row * ncols + j];
  sm[threadIdx.x] = s; __syncthreads();
  for (int w = 128; w > 0; w >>= 1) {
    if (threadIdx.x < w) sm[threadIdx.x] += sm[threadIdx.x + w];
    __syncthreads();
  }
  if (threadIdx.x == 0) out[row] = sm[0];
}

__global__ void colsum_k(const float* __restrict__ M, float* __restrict__ out, int nrows) {
  int col = blockIdx.x * 256 + threadIdx.x;
  if (col >= N_NODES) return;
  float s = 0.0f;
  for (int i = 0; i < nrows; ++i) s += M[(size_t)i * N_NODES + col];
  out[col] = s;
}

__global__ void inv_rsqrt_k(float* d, int n) {
  int i = blockIdx.x * 256 + threadIdx.x;
  if (i < n) {
    float v = d[i];
    d[i] = (v > 0.0f) ? rsqrtf(fmaxf(v, 1e-12f)) : 0.0f;
  }
}

__global__ void conv_a16_k(const float* __restrict__ A32, h16* __restrict__ A16,
                           h16* __restrict__ At16) {
  size_t idx = (size_t)blockIdx.x * 256 + threadIdx.x;
  if (idx >= (size_t)N_NODES * N_NODES) return;
  int i = (int)(idx / N_NODES), j = (int)(idx % N_NODES);
  float v = A32[idx];
  A16[idx] = (h16)v;
  At16[(size_t)j * N_NODES + i] = (h16)v;
}

__global__ void mask_s_k(const float* __restrict__ A32, float* __restrict__ Sout,
                         float* __restrict__ Sin) {
  size_t idx = (size_t)blockIdx.x * 256 + threadIdx.x;
  if (idx >= (size_t)N_NODES * N_NODES) return;
  int i = (int)(idx / N_NODES), j = (int)(idx % N_NODES);
  bool diag = (i == j);
  if (diag || A32[(size_t)i * N_NODES + j] != 0.0f) Sout[idx] = 0.0f;
  if (diag || A32[(size_t)j * N_NODES + i] != 0.0f) Sin[idx]  = 0.0f;
}

// Mcat_sd = [0.35*A_norm | 0.15*S_out_norm]   (3072 x 6144, f16)
__global__ void build_mcat_sd_k(const float* __restrict__ A32, const float* __restrict__ Sout,
                                const float* __restrict__ invOutA, const float* __restrict__ invInA,
                                const float* __restrict__ soOut,   const float* __restrict__ soIn,
                                h16* __restrict__ M) {
  size_t idx = (size_t)blockIdx.x * 256 + threadIdx.x;
  if (idx >= (size_t)N_NODES * KCAT) return;
  int i = (int)(idx / KCAT), j = (int)(idx % KCAT);
  float v;
  if (j < N_NODES) v = 0.35f * invOutA[i] * A32[(size_t)i * N_NODES + j] * invInA[j];
  else { int jj = j - N_NODES; v = 0.15f * soOut[i] * Sout[(size_t)i * N_NODES + jj] * soIn[jj]; }
  M[idx] = (h16)v;
}

// Mcat_ds = [0.35*At_norm | 0.15*S_in_norm]; At_norm[i][j] = invInA[i]*A[j][i]*invOutA[j]
__global__ void build_mcat_ds_k(const float* __restrict__ A32, const float* __restrict__ Sin,
                                const float* __restrict__ invOutA, const float* __restrict__ invInA,
                                const float* __restrict__ siOut,   const float* __restrict__ siIn,
                                h16* __restrict__ M) {
  size_t idx = (size_t)blockIdx.x * 256 + threadIdx.x;
  if (idx >= (size_t)N_NODES * KCAT) return;
  int i = (int)(idx / KCAT), j = (int)(idx % KCAT);
  float v;
  if (j < N_NODES) v = 0.35f * invInA[i] * A32[(size_t)j * N_NODES + i] * invOutA[j];
  else { int jj = j - N_NODES; v = 0.15f * siOut[i] * Sin[(size_t)i * N_NODES + jj] * siIn[jj]; }
  M[idx] = (h16)v;
}

// Xt[f][n] = Xt[f][N+n] = H[n][f]   (H: 3072x512 f32 -> Xt: 512x6144 f16, duplicated)
__global__ void transpose_dup_k(const float* __restrict__ H, h16* __restrict__ Xt) {
  __shared__ float tile[32][33];
  int n0 = blockIdx.x * 32, f0 = blockIdx.y * 32;
  for (int r = threadIdx.y; r < 32; r += 8)
    tile[r][threadIdx.x] = H[(size_t)(n0 + r) * F_DIM + f0 + threadIdx.x];
  __syncthreads();
  for (int r = threadIdx.y; r < 32; r += 8) {
    int f = f0 + r, n = n0 + threadIdx.x;
    h16 v = (h16)tile[threadIdx.x][r];
    Xt[(size_t)f * KCAT + n] = v;
    Xt[(size_t)f * KCAT + N_NODES + n] = v;
  }
}

__global__ void build_wcat_k(const float* __restrict__ Wsd, const float* __restrict__ Wds,
                             h16* __restrict__ Wc, int outReal, int outPad) {
  int idx = blockIdx.x * 256 + threadIdx.x;
  if (idx >= outPad * HK) return;
  int o = idx / HK, j = idx % HK;
  float v = 0.0f;
  if (o < outReal) v = (j < F_DIM) ? Wsd[o * F_DIM + j] : Wds[o * F_DIM + (j - F_DIM)];
  Wc[idx] = (h16)v;
}

__global__ void build_bias_k(const float* __restrict__ bsd, const float* __restrict__ bds,
                             float* __restrict__ bias, int outReal, int outPad) {
  int o = blockIdx.x * 256 + threadIdx.x;
  if (o < outPad) bias[o] = (o < outReal) ? 0.5f * (bsd[o] + bds[o]) : 0.0f;
}

__global__ void hcat_conv_k(const float* __restrict__ Ccat, h16* __restrict__ Hcat) {
  size_t idx = (size_t)blockIdx.x * 256 + threadIdx.x;
  if (idx < (size_t)N_NODES * HK) Hcat[idx] = (h16)Ccat[idx];
}

__global__ void relu_bias_k(const float* __restrict__ O, const float* __restrict__ bias,
                            float* __restrict__ H1) {
  size_t idx = (size_t)blockIdx.x * 256 + threadIdx.x;
  if (idx < (size_t)N_NODES * F_DIM)
    H1[idx] = fmaxf(O[idx] + bias[idx % F_DIM], 0.0f);
}

__global__ void logsoftmax_k(const float* __restrict__ logits, const float* __restrict__ bias,
                             float* __restrict__ out) {
  __shared__ float sm[64];
  int row = blockIdx.x, t = threadIdx.x;
  float v = (t < CLS) ? logits[(size_t)row * CLS_PAD + t] + bias[t] : -INFINITY;
  sm[t] = v; __syncthreads();
  for (int w = 32; w > 0; w >>= 1) {
    if (t < w) sm[t] = fmaxf(sm[t], sm[t + w]);
    __syncthreads();
  }
  float m = sm[0]; __syncthreads();
  float ex = (t < CLS) ? expf(v - m) : 0.0f;
  sm[t] = ex; __syncthreads();
  for (int w = 32; w > 0; w >>= 1) {
    if (t < w) sm[t] += sm[t + w];
    __syncthreads();
  }
  float lse = logf(sm[0]) + m;
  if (t < CLS) out[(size_t)row * CLS + t] = v - lse;
}

// ---------------------------------------------------------------------------
extern "C" void kernel_launch(void* const* d_in, const int* in_sizes, int n_in,
                              void* d_out, int out_size, void* d_ws, size_t ws_size,
                              hipStream_t stream) {
  (void)in_sizes; (void)n_in; (void)out_size; (void)ws_size;
  const float* x    = (const float*)d_in[0];
  const int*   ei   = (const int*)d_in[1];
  const float* Wsd0 = (const float*)d_in[2];
  const float* bsd0 = (const float*)d_in[3];
  const float* Wds0 = (const float*)d_in[4];
  const float* bds0 = (const float*)d_in[5];
  const float* Wsd1 = (const float*)d_in[6];
  const float* bsd1 = (const float*)d_in[7];
  const float* Wds1 = (const float*)d_in[8];
  const float* bds1 = (const float*)d_in[9];
  float* out = (float*)d_out;
  char*  ws  = (char*)d_ws;

  const size_t NN = (size_t)N_NODES * N_NODES;
  // Persistent region (phase 1)
  float* A32  = (float*)(ws + 0);
  h16*   A16  = (h16*)(ws + 37748736);
  h16*   At16 = (h16*)(ws + 56623104);
  float* Sout = (float*)(ws + 75497472);
  float* Sin  = (float*)(ws + 113246208);
  h16*   Msd  = (h16*)(ws + 150994944);
  h16*   Mds  = (h16*)(ws + 188743680);
  float* smallp  = (float*)(ws + 226492416);
  float* invOutA = smallp + 0 * N_NODES;
  float* invInA  = smallp + 1 * N_NODES;
  float* soOut   = smallp + 2 * N_NODES;
  float* soIn    = smallp + 3 * N_NODES;
  float* siOut   = smallp + 4 * N_NODES;
  float* siIn    = smallp + 5 * N_NODES;
  // Reuse region (valid after Mcat matrices are built; overlaps dead A/S)
  h16*   Xt    = (h16*)(ws + 0);
  float* Ccat  = (float*)(ws + 8388608);
  h16*   Hcat  = (h16*)(ws + 25165824);
  float* O1    = (float*)(ws + 33554432);
  float* H1    = (float*)(ws + 41943040);
  float* Logit = (float*)(ws + 50331648);
  h16*   Wc1   = (h16*)(ws + 52428800);
  h16*   Wc2   = (h16*)(ws + 54525952);
  float* bias1 = (float*)(ws + 55574528);
  float* bias2 = (float*)(ws + 55578624);

  dim3 B(256);
  // 1) build A
  zero_f32<<<dim3((unsigned)((NN + 255) / 256)), B, 0, stream>>>(A32, NN);
  scatter_edges<<<dim3((NE + 255) / 256), B, 0, stream>>>(ei, A32);
  // 2) A degrees -> inverse sqrt
  rowsum_k<<<dim3(N_NODES), B, 0, stream>>>(A32, invOutA, N_NODES);
  colsum_k<<<dim3(N_NODES / 256), B, 0, stream>>>(A32, invInA, N_NODES);
  inv_rsqrt_k<<<dim3(N_NODES / 256), B, 0, stream>>>(invOutA, N_NODES);
  inv_rsqrt_k<<<dim3(N_NODES / 256), B, 0, stream>>>(invInA, N_NODES);
  // 3) raw A/At in f16 (exact: small integer counts)
  conv_a16_k<<<dim3((unsigned)((NN + 255) / 256)), B, 0, stream>>>(A32, A16, At16);
  // 4) second-order: Sout = A@A^T, Sin = At@At^T  (f16 WMMA, f32 accumulate)
  dim3 gS(N_NODES / 128, N_NODES / 64);
  gemm_tn_f16f32<<<gS, B, 0, stream>>>(A16,  A16,  Sout, N_NODES, N_NODES, N_NODES, N_NODES);
  gemm_tn_f16f32<<<gS, B, 0, stream>>>(At16, At16, Sin,  N_NODES, N_NODES, N_NODES, N_NODES);
  // 5) zero where first-order edge exists / diagonal
  mask_s_k<<<dim3((unsigned)((NN + 255) / 256)), B, 0, stream>>>(A32, Sout, Sin);
  // 6) S degrees -> inverse sqrt
  rowsum_k<<<dim3(N_NODES), B, 0, stream>>>(Sout, soOut, N_NODES);
  colsum_k<<<dim3(N_NODES / 256), B, 0, stream>>>(Sout, soIn, N_NODES);
  rowsum_k<<<dim3(N_NODES), B, 0, stream>>>(Sin, siOut, N_NODES);
  colsum_k<<<dim3(N_NODES / 256), B, 0, stream>>>(Sin, siIn, N_NODES);
  inv_rsqrt_k<<<dim3(N_NODES / 256), B, 0, stream>>>(soOut, N_NODES);
  inv_rsqrt_k<<<dim3(N_NODES / 256), B, 0, stream>>>(soIn, N_NODES);
  inv_rsqrt_k<<<dim3(N_NODES / 256), B, 0, stream>>>(siOut, N_NODES);
  inv_rsqrt_k<<<dim3(N_NODES / 256), B, 0, stream>>>(siIn, N_NODES);
  // 7) concatenated, pre-scaled propagation matrices (f16)
  unsigned gM = (unsigned)(((size_t)N_NODES * KCAT + 255) / 256);
  build_mcat_sd_k<<<dim3(gM), B, 0, stream>>>(A32, Sout, invOutA, invInA, soOut, soIn, Msd);
  build_mcat_ds_k<<<dim3(gM), B, 0, stream>>>(A32, Sin,  invOutA, invInA, siOut, siIn, Mds);
  // 8) weights / biases
  build_wcat_k<<<dim3((F_DIM * HK + 255) / 256), B, 0, stream>>>(Wsd0, Wds0, Wc1, F_DIM, F_DIM);
  build_wcat_k<<<dim3((CLS_PAD * HK + 255) / 256), B, 0, stream>>>(Wsd1, Wds1, Wc2, CLS, CLS_PAD);
  build_bias_k<<<dim3(2), B, 0, stream>>>(bsd0, bds0, bias1, F_DIM, F_DIM);
  build_bias_k<<<dim3(1), B, 0, stream>>>(bsd1, bds1, bias2, CLS, CLS_PAD);

  dim3 Bt(32, 8);
  dim3 gT(N_NODES / 32, F_DIM / 32);
  dim3 gU(N_NODES / 128, F_DIM / 64);
  dim3 gF1(N_NODES / 128, F_DIM / 64);
  dim3 gF2(N_NODES / 128, CLS_PAD / 64);
  unsigned gH = (unsigned)(((size_t)N_NODES * HK + 255) / 256);
  unsigned gR = (unsigned)(((size_t)N_NODES * F_DIM + 255) / 256);

  // ---- layer 1 ----
  transpose_dup_k<<<gT, Bt, 0, stream>>>(x, Xt);
  gemm_tn_f16f32<<<gU, B, 0, stream>>>(Msd, Xt, Ccat,       KCAT, KCAT, HK, KCAT);
  gemm_tn_f16f32<<<gU, B, 0, stream>>>(Mds, Xt, Ccat + 512, KCAT, KCAT, HK, KCAT);
  hcat_conv_k<<<dim3(gH), B, 0, stream>>>(Ccat, Hcat);
  gemm_tn_f16f32<<<gF1, B, 0, stream>>>(Hcat, Wc1, O1, HK, HK, F_DIM, HK);
  relu_bias_k<<<dim3(gR), B, 0, stream>>>(O1, bias1, H1);

  // ---- layer 2 ----
  transpose_dup_k<<<gT, Bt, 0, stream>>>(H1, Xt);
  gemm_tn_f16f32<<<gU, B, 0, stream>>>(Msd, Xt, Ccat,       KCAT, KCAT, HK, KCAT);
  gemm_tn_f16f32<<<gU, B, 0, stream>>>(Mds, Xt, Ccat + 512, KCAT, KCAT, HK, KCAT);
  hcat_conv_k<<<dim3(gH), B, 0, stream>>>(Ccat, Hcat);
  gemm_tn_f16f32<<<gF2, B, 0, stream>>>(Hcat, Wc2, Logit, HK, HK, CLS_PAD, HK);
  logsoftmax_k<<<dim3(N_NODES), dim3(64), 0, stream>>>(Logit, bias2, out);
}
